// AttentionWithRoPE_15779709845740
// MI455X (gfx1250) — compile-verified, matrix operand order checked by reading, not run on hardware
//
#include <hip/hip_runtime.h>

// ---------------------------------------------------------------------------
// AttentionWithRoPE for MI455X (gfx1250, wave32, WMMA).
// B=2, S=2048, HID=2048, H=HKV=16, D=128.  QKVO = (H+2*HKV)*D = 6144.
//
// Workspace layout (needs 64 MiB):
//   [0, 48 MiB)  : qkv bf16, [4096 tokens][6144]   (q | k | v per token)
//   [48, 64 MiB) : attn out bf16, [4096 tokens][2048]
// ---------------------------------------------------------------------------

typedef __attribute__((ext_vector_type(16))) __bf16 bf16x16;
typedef __attribute__((ext_vector_type(8)))  __bf16 bf16x8;
typedef __attribute__((ext_vector_type(4)))  __bf16 bf16x4;
typedef __attribute__((ext_vector_type(8)))  float  f32x8;

#define NTOK   4096
#define SEQ    2048
#define HID    2048
#define NHEAD  16
#define HDIM   128
#define QKVO   6144

// Build a 16-bit A/B-layout WMMA fragment from a K-major row.
// Per ISA 7.12.2: lanes 0-15 hold K = {0..7, 16..23}, lanes 16-31 hold
// K = {8..15, 24..31} (pairs packed per VGPR). So a fragment is two
// contiguous 16-byte chunks at row+kb8 and row+kb8+16.
__device__ __forceinline__ bf16x16 make_frag(const __bf16* rowbase, int kb8) {
  bf16x8 lo = *(const bf16x8*)(rowbase + kb8);
  bf16x8 hi = *(const bf16x8*)(rowbase + kb8 + 16);
  return __builtin_shufflevector(lo, hi, 0,1,2,3,4,5,6,7,8,9,10,11,12,13,14,15);
}

__device__ __forceinline__ f32x8 wmma_bf16(bf16x16 a, bf16x16 b, f32x8 c) {
  // (neg_a, A, neg_b, B, c_mod, C, reuse_a, reuse_b)
  return __builtin_amdgcn_wmma_f32_16x16x32_bf16(false, a, false, b,
                                                 (short)0, c, false, false);
}

// ---------------------------------------------------------------------------
// Tiled GEMM:  C[M][N] = A[M][K] * Bt[N][K]^T  (Bt is N-major, K contiguous)
//   A: f32 or bf16 (template), Bt: f32, C: f32 or bf16 (template).
// WG = 256 threads (8 waves), 128(M) x 256(N) output tile, K-step 32.
// Waves laid out 2(M) x 4(N); each wave owns 64x64 = 4x4 16x16 WMMA tiles:
// 16 WMMAs per 16 ds_load_b128 per K-step (vs 8:12 with 64x32 waves).
// Global->register->LDS software pipeline: K-step k+1 is fetched into
// staging VGPRs while the WMMAs of K-step k run off LDS.
// ---------------------------------------------------------------------------
template<bool A_BF16, bool OUT_BF16>
__global__ __launch_bounds__(256)
void gemm_wmma(const void* __restrict__ Aptr, const float* __restrict__ Bt,
               void* __restrict__ Cptr, int M, int N, int K)
{
  __shared__ __bf16 As[128 * 40];   // rows padded to 40 bf16 (80 B, 16B-mult)
  __shared__ __bf16 Bs[256 * 40];

  const int tid  = threadIdx.x;
  const int lane = tid & 31;
  const int w    = tid >> 5;
  const int wm   = w & 1;          // 0..1 -> 64-row slab
  const int wn   = w >> 1;         // 0..3 -> 64-col slab
  const int hf   = lane >> 4;      // half of wave (0/1)
  const int l16  = lane & 15;
  const int kb8  = hf * 8;
  const int m0   = blockIdx.y * 128;
  const int n0   = blockIdx.x * 256;

  f32x8 acc[4][4] = {};

  // ---- staging registers for the global->LDS pipeline ----
  float4 aF[4];      // A tile, f32 source (128x32 / 256 thr = 4 float4)
  bf16x8 aB[2];      // A tile, bf16 source (128x32 / 256 thr = 2 bf16x8)
  float4 bF[8];      // B tile, f32 source (256x32 / 256 thr = 8 float4)

  auto load_stage = [&](int kt) {
    if constexpr (A_BF16) {
      const __bf16* A = (const __bf16*)Aptr;
      #pragma unroll
      for (int p = 0; p < 2; ++p) {
        int linear = tid + 256 * p;            // 0..511
        int row = linear >> 2;
        int kq  = (linear & 3) * 8;
        aB[p] = *(const bf16x8*)(A + (size_t)(m0 + row) * K + kt + kq);
      }
    } else {
      const float* A = (const float*)Aptr;
      #pragma unroll
      for (int p = 0; p < 4; ++p) {
        int linear = tid + 256 * p;            // 0..1023
        int row = linear >> 3;
        int kq  = (linear & 7) * 4;
        aF[p] = *(const float4*)(A + (size_t)(m0 + row) * K + kt + kq);
      }
    }
    #pragma unroll
    for (int p = 0; p < 8; ++p) {
      int linear = tid + 256 * p;              // 0..2047
      int row = linear >> 3;
      int kq  = (linear & 7) * 4;
      bF[p] = *(const float4*)(Bt + (size_t)(n0 + row) * K + kt + kq);
    }
  };

  auto store_stage = [&]() {
    if constexpr (A_BF16) {
      #pragma unroll
      for (int p = 0; p < 2; ++p) {
        int linear = tid + 256 * p;
        int row = linear >> 2;
        int kq  = (linear & 3) * 8;
        *(bf16x8*)(As + row * 40 + kq) = aB[p];
      }
    } else {
      #pragma unroll
      for (int p = 0; p < 4; ++p) {
        int linear = tid + 256 * p;
        int row = linear >> 3;
        int kq  = (linear & 7) * 4;
        float4 v = aF[p];
        bf16x4 bv = { (__bf16)v.x, (__bf16)v.y, (__bf16)v.z, (__bf16)v.w };
        *(bf16x4*)(As + row * 40 + kq) = bv;
      }
    }
    #pragma unroll
    for (int p = 0; p < 8; ++p) {
      int linear = tid + 256 * p;
      int row = linear >> 3;
      int kq  = (linear & 7) * 4;
      float4 v = bF[p];
      bf16x4 bv = { (__bf16)v.x, (__bf16)v.y, (__bf16)v.z, (__bf16)v.w };
      *(bf16x4*)(Bs + row * 40 + kq) = bv;
    }
  };

  load_stage(0);
  for (int kt = 0; kt < K; kt += 32) {
    store_stage();
    __syncthreads();
    if (kt + 32 < K) load_stage(kt + 32);   // prefetch under the WMMAs

    bf16x16 af[4], bfr[4];
    #pragma unroll
    for (int mi = 0; mi < 4; ++mi)
      af[mi] = make_frag(As + (wm * 64 + mi * 16 + l16) * 40, kb8);
    #pragma unroll
    for (int ni = 0; ni < 4; ++ni)
      bfr[ni] = make_frag(Bs + (wn * 64 + ni * 16 + l16) * 40, kb8);
    #pragma unroll
    for (int mi = 0; mi < 4; ++mi)
      #pragma unroll
      for (int ni = 0; ni < 4; ++ni)
        acc[mi][ni] = wmma_bf16(af[mi], bfr[ni], acc[mi][ni]);
    __syncthreads();   // LDS consumed; next store_stage may overwrite
  }

  // ---- epilogue: C layout = VGPR r -> row (r + 8*hf), lane&15 -> col ----
  #pragma unroll
  for (int mi = 0; mi < 4; ++mi) {
    #pragma unroll
    for (int ni = 0; ni < 4; ++ni) {
      #pragma unroll
      for (int r = 0; r < 8; ++r) {
        int row = m0 + wm * 64 + mi * 16 + r + 8 * hf;
        int col = n0 + wn * 64 + ni * 16 + l16;
        float val = acc[mi][ni][r];
        if constexpr (OUT_BF16)
          ((__bf16*)Cptr)[(size_t)row * N + col] = (__bf16)val;
        else
          ((float*)Cptr)[(size_t)row * N + col] = val;
      }
    }
  }
}

// ---------------------------------------------------------------------------
// RoPE, in place on bf16 qkv (q and k heads). Thread handles (d, d+64) pair.
// cos/sin tables are [S][128] with duplicated halves: tab[s][d]==tab[s][d+64].
// ---------------------------------------------------------------------------
__global__ __launch_bounds__(256)
void rope_kernel(void* qkv_v, const float* __restrict__ cosT,
                 const float* __restrict__ sinT)
{
  __bf16* qkv = (__bf16*)qkv_v;
  size_t idx = (size_t)blockIdx.x * blockDim.x + threadIdx.x;
  int d   = (int)(idx & 63);
  int hh  = (int)((idx >> 6) & 31);     // 0..15 = q heads, 16..31 = k heads
  int tok = (int)(idx >> 11);           // 0..4095
  int s   = tok & (SEQ - 1);
  size_t base = (size_t)tok * QKVO + hh * HDIM;   // q at hh*128, k at same
  float x1 = (float)qkv[base + d];
  float x2 = (float)qkv[base + d + 64];
  float c  = cosT[s * HDIM + d];
  float sn = sinT[s * HDIM + d];
  qkv[base + d]      = (__bf16)(x1 * c - x2 * sn);
  qkv[base + d + 64] = (__bf16)(x2 * c + x1 * sn);
}

// ---------------------------------------------------------------------------
// Causal flash attention. Grid = (B*H, S/64). Block = 128 threads (4 waves),
// wave w owns query rows [q0 + 16w, q0 + 16w + 16). Key blocks of 64.
// ---------------------------------------------------------------------------
__global__ __launch_bounds__(128)
void flash_attn(const void* __restrict__ qkv_v, void* __restrict__ attn_v)
{
  const __bf16* qkv = (const __bf16*)qkv_v;
  __bf16* attn = (__bf16*)attn_v;

  __shared__ __bf16 Ks[64 * 136];   // [key][d], row stride 136 (272 B)
  __shared__ __bf16 Vt[128 * 72];   // [d][key], transposed V, stride 72
  __shared__ __bf16 Ps[4 * 16 * 72];// per-wave P tile [16 q][64 key], stride 72

  const int tid  = threadIdx.x;
  const int lane = tid & 31;
  const int w    = tid >> 5;
  const int hf   = lane >> 4;
  const int l16  = lane & 15;
  const int kb8  = hf * 8;
  const int b    = blockIdx.x >> 4;
  const int h    = blockIdx.x & 15;
  const int q0   = blockIdx.y * 64;
  const int qw   = q0 + w * 16;

  // Q fragments held in registers for the whole kernel (global is K-major,
  // so Q loads directly in A-fragment layout).
  bf16x16 qf[4];
  {
    const __bf16* gq = qkv + (size_t)(b * SEQ + qw + l16) * QKVO + h * HDIM;
    #pragma unroll
    for (int kt = 0; kt < 4; ++kt) qf[kt] = make_frag(gq + kt * 32, kb8);
  }

  f32x8 oacc[8] = {};
  float m_i[8], l_i[8];
  #pragma unroll
  for (int r = 0; r < 8; ++r) { m_i[r] = -1e30f; l_i[r] = 0.f; }

  const int kb_end = (q0 + 63) >> 6;             // inclusive
  for (int kb = 0; kb <= kb_end; ++kb) {
    const int k0 = kb * 64;

    // ---- cooperative stage: K tile as-is, V tile transposed ----
    #pragma unroll
    for (int i = 0; i < 8; ++i) {
      int idx = tid + 128 * i;                   // 0..1023
      int row = idx >> 4;                        // key 0..63
      int c16 = idx & 15;                        // 8-elem chunk over d
      size_t tok = (size_t)(b * SEQ + k0 + row) * QKVO;
      bf16x8 kv = *(const bf16x8*)(qkv + tok + (NHEAD + h) * HDIM + c16 * 8);
      *(bf16x8*)(Ks + row * 136 + c16 * 8) = kv;
      bf16x8 vv = *(const bf16x8*)(qkv + tok + (2 * NHEAD + h) * HDIM + c16 * 8);
      #pragma unroll
      for (int j = 0; j < 8; ++j) Vt[(c16 * 8 + j) * 72 + row] = vv[j];
    }
    __syncthreads();

    // ---- scores: 16 q x 64 key, K-dim = 128 (4 WMMA steps) ----
    f32x8 sc[4] = {};
    #pragma unroll
    for (int kt = 0; kt < 4; ++kt) {
      #pragma unroll
      for (int t = 0; t < 4; ++t) {
        bf16x16 kfrag = make_frag(Ks + (t * 16 + l16) * 136 + kt * 32, kb8);
        sc[t] = wmma_bf16(qf[kt], kfrag, sc[t]);
      }
    }

    // ---- scale + causal mask + row max (row = 16 lanes of one half) ----
    const float scale = 0.08838834764831845f;    // 1/sqrt(128)
    float mnewv[8];
    #pragma unroll
    for (int r = 0; r < 8; ++r) {
      int qrow = qw + r + 8 * hf;
      float v = -1e30f;
      #pragma unroll
      for (int t = 0; t < 4; ++t) {
        int key = k0 + t * 16 + l16;
        float s = sc[t][r] * scale;
        if (key > qrow) s = -1e30f;
        sc[t][r] = s;
        v = fmaxf(v, s);
      }
      #pragma unroll
      for (int msk = 8; msk >= 1; msk >>= 1) v = fmaxf(v, __shfl_xor(v, msk));
      mnewv[r] = fmaxf(m_i[r], v);
    }

    // ---- exp, row sum, rescale accumulators, write P (bf16) to LDS ----
    #pragma unroll
    for (int r = 0; r < 8; ++r) {
      float mnew  = mnewv[r];
      float alpha = __expf(m_i[r] - mnew);
      float sum = 0.f;
      #pragma unroll
      for (int t = 0; t < 4; ++t) {
        float p = __expf(sc[t][r] - mnew);
        sum += p;
        Ps[(w * 16 + r + 8 * hf) * 72 + t * 16 + l16] = (__bf16)p;
      }
      #pragma unroll
      for (int msk = 8; msk >= 1; msk >>= 1) sum += __shfl_xor(sum, msk);
      l_i[r] = l_i[r] * alpha + sum;
      m_i[r] = mnew;
      #pragma unroll
      for (int dt = 0; dt < 8; ++dt) oacc[dt][r] *= alpha;
    }

    // ---- O += P (16x64) x V (64x128): 2 K-steps x 8 d-tiles ----
    #pragma unroll
    for (int kt2 = 0; kt2 < 2; ++kt2) {
      bf16x16 pf = make_frag(Ps + (w * 16 + l16) * 72 + kt2 * 32, kb8);
      #pragma unroll
      for (int dt = 0; dt < 8; ++dt) {
        bf16x16 vfrag = make_frag(Vt + (dt * 16 + l16) * 72 + kt2 * 32, kb8);
        oacc[dt] = wmma_bf16(pf, vfrag, oacc[dt]);
      }
    }
    __syncthreads();   // protect Ks/Vt before next block's stage
  }

  // ---- normalize and write attn output (token-major, [4096][2048]) ----
  #pragma unroll
  for (int dt = 0; dt < 8; ++dt) {
    #pragma unroll
    for (int r = 0; r < 8; ++r) {
      float val = oacc[dt][r] / l_i[r];
      size_t o = (size_t)(b * SEQ + qw + r + 8 * hf) * HID
               + h * HDIM + dt * 16 + l16;
      attn[o] = (__bf16)val;
    }
  }
}

// ---------------------------------------------------------------------------
extern "C" void kernel_launch(void* const* d_in, const int* in_sizes, int n_in,
                              void* d_out, int out_size, void* d_ws, size_t ws_size,
                              hipStream_t stream) {
  const float* hidden = (const float*)d_in[0];   // [2,2048,2048]
  const float* cosT   = (const float*)d_in[1];   // [2048,128]
  const float* sinT   = (const float*)d_in[2];   // [2048,128]
  const float* w_qkv  = (const float*)d_in[3];   // [6144,2048]
  const float* w_o    = (const float*)d_in[4];   // [2048,2048]
  float* out = (float*)d_out;                    // [2,2048,2048] f32

  void* qkv_ws  = d_ws;                                           // 48 MiB bf16
  void* attn_ws = (void*)((char*)d_ws + (size_t)NTOK * QKVO * 2); // 16 MiB

  // 1) QKV projection: [4096,2048] x [6144,2048]^T -> bf16 qkv
  gemm_wmma<false, true><<<dim3(QKVO / 256, NTOK / 128), 256, 0, stream>>>(
      hidden, w_qkv, qkv_ws, NTOK, QKVO, HID);

  // 2) RoPE in place on q,k (4096 tok * 32 heads * 64 pairs = 8388608 threads)
  rope_kernel<<<(NTOK * 32 * 64) / 256, 256, 0, stream>>>(qkv_ws, cosT, sinT);

  // 3) Causal flash attention -> bf16 attn
  flash_attn<<<dim3(2 * NHEAD, SEQ / 64), 128, 0, stream>>>(qkv_ws, attn_ws);

  // 4) Output projection: [4096,2048] x [2048,2048]^T -> f32 out
  gemm_wmma<true, false><<<dim3(HID / 256, NTOK / 128), 256, 0, stream>>>(
      attn_ws, w_o, out, NTOK, HID, HID);
}
